// KPConv_81604378624786
// MI455X (gfx1250) — matrix operand order repaired and checked by reading, not compiled, not run on hardware
//
#include <hip/hip_runtime.h>
#include <hip/hip_bf16.h>

// KPConv fused kernel for gfx1250 (MI455X).
//  N=65536 query pts, Ns=65536 support pts, H=26 neighbors, PAD_H=28,
//  K=10 kernel points, Cin=64, Cout=128, KP_EXTENT=1.2.
//
// One wave32 per 16-query-point tile, fully fused:
//   step5 (all_w @ neighb_x): 16 pts x 4 WMMA f32_16x16x32_f16
//   step6 ([16x640] @ [640x128]): 20 ksteps x 8 cout tiles = 160 WMMA
// fp16 operands, fp32 accumulation. All memory ops are branch-free:
// padded neighbor slots gather from a shadow zero row of the feature table.

typedef __attribute__((ext_vector_type(16))) _Float16 v16h;
typedef __attribute__((ext_vector_type(8)))  float    v8f;

#define HN   26
#define PADH 28
#define KPN  10
#define CIN  64
#define COUT 128
#define FLATK 640           // KPN*CIN
#define KSTEPS 20           // FLATK/32

// ---------------- prep kernels ----------------

// x [Ns*64] f32 -> f16 table with an extra all-zero shadow row at index Ns.
__global__ void cvt_x_f16(const float* __restrict__ x, _Float16* __restrict__ X16,
                          int n, int ntot) {
    int i = blockIdx.x * blockDim.x + threadIdx.x;
    if (i < ntot) X16[i] = (i < n) ? (_Float16)x[i] : (_Float16)0.f;
}

// weights [K,Cin,Cout] f32 -> fp16, pre-swizzled into per-WMMA-fragment blocks:
//   block(ks in [0,20), j in [0,8)) = B [32 K][16 cols], stored column-major
//   (16 cols x 32 contiguous K halves) so a lane loads its two 16B K runs (table order) with two b128s.
__global__ void swz_w_f16(const float* __restrict__ w, _Float16* __restrict__ Wsw) {
    int i = blockIdx.x * blockDim.x + threadIdx.x;    // over 640*128
    if (i < FLATK * COUT) {
        int t = i / COUT, n = i % COUT;               // t = k*64+cin (flat K), n = cout
        int ks = t >> 5, kk = t & 31, j = n >> 4, col = n & 15;
        Wsw[(((ks * 8 + j) * 16 + col) << 5) + kk] = (_Float16)w[t * COUT + n];
    }
}

// ---------------- main fused kernel ----------------

__global__ __launch_bounds__(32) void kpconv_main(
    const float* __restrict__ q_pts, const float* __restrict__ s_pts,
    const float* __restrict__ genW,  const float* __restrict__ genB,
    const int* __restrict__ nbi,
    const _Float16* __restrict__ X16, const _Float16* __restrict__ Wsw,
    float* __restrict__ out, int zeroOff)
{
    __shared__ float s_nbr[16][32][4];                 // neighbor deltas (h 26,27 = -1 pad)
    __shared__ float s_kp [16][16][4];                 // kernel points (rows 10..15 zeroed)
    __shared__ __align__(16) int s_off[16][32];        // byte offset of gathered X16 row
    __shared__ __align__(16) _Float16 s_A7[16][FLATK]; // weighted [pt][k*64+cin] (step6 A)
    __shared__ _Float16 s_dump[32];                    // dump slot for k>=10 D rows

    const int lane   = threadIdx.x;
    const int n0     = blockIdx.x * 16;                // first query point of tile
    const int col    = lane & 15;
    const bool hiH   = lane >= 16;
    const int hb     = hiH ? 8 : 0;

    // ---- Phase A: gather neighbor coords (divergence-free: 416 = 13 full waves) ----
    for (int i = lane; i < 16 * HN; i += 32) {
        int p = i / HN, h = i - p * HN;
        int idx = nbi[(n0 + p) * HN + h];
        const float* sp = s_pts + (size_t)idx * 3;
        const float* qp = q_pts + (size_t)(n0 + p) * 3;
        s_nbr[p][h][0] = sp[0] - qp[0];
        s_nbr[p][h][1] = sp[1] - qp[1];
        s_nbr[p][h][2] = sp[2] - qp[2];
        s_nbr[p][h][3] = 0.f;
        s_off[p][h] = idx * (CIN * 2);
    }
    // padding h=26..31: generator pad (-1) for 26,27, zero beyond; gathers hit zero row
    for (int i = lane; i < 16 * 6; i += 32) {
        int p = i / 6, h = HN + (i - (i / 6) * 6);
        float v = (h < PADH) ? -1.0f : 0.0f;
        s_nbr[p][h][0] = v; s_nbr[p][h][1] = v; s_nbr[p][h][2] = v; s_nbr[p][h][3] = 0.f;
        s_off[p][h] = zeroOff;
    }
    __syncthreads();

    // ---- Phase B: kernel points kp = genW @ padded + genB (480 = 15 full waves) ----
    for (int i = lane; i < 16 * (KPN * 3); i += 32) {
        int p = i / (KPN * 3), o = i - p * (KPN * 3);
        float acc = genB[o];
        const float* wr = genW + o * (PADH * 3);
        for (int h = 0; h < PADH; ++h)
            for (int d = 0; d < 3; ++d)
                acc = fmaf(s_nbr[p][h][d], wr[h * 3 + d], acc);
        s_kp[p][o / 3][o % 3] = acc;
    }
    for (int i = lane; i < 16 * 6; i += 32) {          // zero unused kp rows 10..15
        int p = i / 6, kr = KPN + (i - (i / 6) * 6);
        s_kp[p][kr][0] = 0.f; s_kp[p][kr][1] = 0.f; s_kp[p][kr][2] = 0.f; s_kp[p][kr][3] = 0.f;
    }
    __syncthreads();

    // ---- Phase C: per point, all_w A fragment (in-register, ISA A layout),
    //      gathered fp16 B fragments -> 4 WMMAs -> weighted rows into s_A7 ----
    const int krow  = col;                             // A rows = kernel-point index
    const int hbase = hiH ? 8 : 0;                     // this lane's K(h) base (table order)
    for (int p = 0; p < 16; ++p) {
        // A: lane<16 holds row m=lane, K(h) in {0..7,16..23};
        //    lane>=16 row m=lane-16, K(h) in {8..15,24..31}. Invalid -> 0 via cndmask.
        union { v16h v; _Float16 h[16]; } A6;
        float kx = s_kp[p][krow][0], ky = s_kp[p][krow][1], kz = s_kp[p][krow][2];
        #pragma unroll
        for (int v = 0; v < 8; ++v) {
            int h0 = ((v < 4) ? 2 * v : 2 * v + 8) + hb;
            #pragma unroll
            for (int q = 0; q < 2; ++q) {
                int hh = h0 + q;
                float dx = s_nbr[p][hh][0] - kx;
                float dy = s_nbr[p][hh][1] - ky;
                float dz = s_nbr[p][hh][2] - kz;
                float d  = sqrtf(dx * dx + dy * dy + dz * dz);
                float wv = fmaxf(1.0f - d * (1.0f / 1.2f), 0.0f);
                wv = (krow < KPN && hh < HN) ? wv : 0.0f;
                A6.h[2 * v + q] = (_Float16)wv;
            }
        }
        // this lane's 16 feature-row byte offsets (always valid; pads hit zero row)
        union { int4 q[4]; int s[16]; } offs;
        const int4* op = (const int4*)&s_off[p][hbase];
        offs.q[0] = op[0]; offs.q[1] = op[1]; offs.q[2] = op[4]; offs.q[3] = op[5];

        const char* xb = (const char*)X16;
        #pragma unroll
        for (int c = 0; c < 4; ++c) {                  // Cin tiles of 16
            union { v16h v; _Float16 h[16]; } B;
            int cb = (16 * c + col) * 2;               // byte offset within feature row
            #pragma unroll
            for (int v = 0; v < 8; ++v) {              // branch-free 2B gathers
                B.h[2 * v]     = *(const _Float16*)(xb + offs.s[2 * v]     + cb);
                B.h[2 * v + 1] = *(const _Float16*)(xb + offs.s[2 * v + 1] + cb);
            }
            v8f d = {};
            d = __builtin_amdgcn_wmma_f32_16x16x32_f16(false, A6.v, false, B.v,
                                                       (short)0, d, false, false);
            // D: VGPR v -> row (v + 8*hiH), col = lane&15. k>=10 rows -> dump slot
            #pragma unroll
            for (int v = 0; v < 8; ++v) {
                int kr = v + hb;
                _Float16* dst = (kr < KPN) ? &s_A7[p][kr * CIN + 16 * c + col]
                                           : &s_dump[lane];
                *dst = (_Float16)d[v];
            }
        }
    }
    __syncthreads();

    // ---- Phase D: out tile [16 pts x 128] = s_A7 [16x640] @ Wsw [640x128] ----
    v8f acc[8];
    #pragma unroll
    for (int j = 0; j < 8; ++j) acc[j] = (v8f){};
    const _Float16* a7row = &s_A7[col][0];             // A rows = points
    #pragma unroll 2
    for (int ks = 0; ks < KSTEPS; ++ks) {
        if (ks + 1 < KSTEPS)                           // prefetch next kstep's B (8 KB)
            __builtin_prefetch(Wsw + (size_t)(ks + 1) * 8 * 512 + lane * 128, 0, 0);
        union { v16h v; uint4 q[2]; } A;
        const uint4* ap = (const uint4*)(a7row + 32 * ks + (hiH ? 8 : 0));
        A.q[0] = ap[0];                                // K {0..7} / {8..15}
        A.q[1] = ap[2];                                // K {16..23} / {24..31}
        #pragma unroll
        for (int j = 0; j < 8; ++j) {
            union { v16h v; uint4 q[2]; } B;
            const uint4* bp = (const uint4*)(Wsw + (((size_t)ks * 8 + j) * 16 + col) * 32
                                                 + (hiH ? 8 : 0));
            B.q[0] = bp[0]; B.q[1] = bp[2];
            acc[j] = __builtin_amdgcn_wmma_f32_16x16x32_f16(false, A.v, false, B.v,
                                                            (short)0, acc[j], false, false);
        }
    }

    // ---- Phase E: store out tile (D rows = points, cols = cout) ----
    #pragma unroll
    for (int j = 0; j < 8; ++j)
        #pragma unroll
        for (int v = 0; v < 8; ++v)
            out[(size_t)(n0 + v + hb) * COUT + 16 * j + col] = acc[j][v];
}

// ---------------- host launcher ----------------

extern "C" void kernel_launch(void* const* d_in, const int* in_sizes, int n_in,
                              void* d_out, int out_size, void* d_ws, size_t ws_size,
                              hipStream_t stream) {
    const float* q_pts = (const float*)d_in[0];
    const float* s_pts = (const float*)d_in[1];
    const float* x     = (const float*)d_in[2];
    const float* genW  = (const float*)d_in[3];
    const float* genB  = (const float*)d_in[4];
    const float* wts   = (const float*)d_in[5];
    const int*   nbi   = (const int*)d_in[6];
    float* out = (float*)d_out;

    const int N  = in_sizes[0] / 3;
    const int Ns = in_sizes[1] / 3;

    _Float16* X16 = (_Float16*)d_ws;                   // (Ns+1)*64 f16 (8 MB + zero row)
    _Float16* Wsw = (_Float16*)((char*)d_ws + (size_t)(Ns + 1) * CIN * 2);  // 160 KB

    const int nx   = Ns * CIN;
    const int nxt  = (Ns + 1) * CIN;                   // includes shadow zero row
    cvt_x_f16<<<(nxt + 255) / 256, 256, 0, stream>>>(x, X16, nx, nxt);
    swz_w_f16<<<(FLATK * COUT + 255) / 256, 256, 0, stream>>>(wts, Wsw);
    kpconv_main<<<N / 16, 32, 0, stream>>>(q_pts, s_pts, genW, genB, nbi, X16, Wsw, out,
                                           Ns * CIN * 2);
}
